// BasicAttentionLayer_21586505630423
// MI455X (gfx1250) — matrix operand order repaired and checked by reading, not compiled, not run on hardware
//
#include <hip/hip_runtime.h>
#include <hip/hip_bf16.h>
#include <math.h>

// ---------------------------------------------------------------------------
// MHA layer for MI455X (gfx1250, wave32, WMMA).
//   B=4, S=1024, E=1024, H=16, d=64.
// fp32 -> bf16 once (native convert when available), all GEMMs via
// v_wmma_f32_16x16x32_bf16 (fp32 accumulate), flash-style online softmax in
// the C-fragment layout, K tiles staged to LDS via GLOBAL_LOAD_ASYNC_TO_LDS.
// Workspace: Q/K/V bf16 = 3 * 4096*1024*2B = 24 MB of d_ws.
// ---------------------------------------------------------------------------

#define EMBED 1024
#define HEADS 16
#define HD    64
#define BATCH 4
#define SEQ   1024

#define USE_ASYNC_LDS 1

typedef __attribute__((ext_vector_type(16))) __bf16 v16bf;
typedef __attribute__((ext_vector_type(8)))  float  v8f;

union FragU { uint4 q[2]; v16bf v; };

// fp32 -> bf16 (RNE). Prefer the packed-cvt builtin; else rely on the C-level
// __bf16 cast (fptrunc f32->bf16, native v_cvt on gfx1250); the backend picks
// the best lowering either way.
__device__ __forceinline__ unsigned short bf16_one(float f) {
#if __has_builtin(__builtin_amdgcn_cvt_pk_bf16_f32)
  auto pk = __builtin_amdgcn_cvt_pk_bf16_f32(f, 0.0f);
  unsigned u;
  __builtin_memcpy(&u, &pk, sizeof(u));
  return (unsigned short)(u & 0xFFFFu);
#else
  __bf16 h = (__bf16)f;
  return __builtin_bit_cast(unsigned short, h);
#endif
}

__device__ __forceinline__ unsigned pack2_bf16(float lo, float hi) {
#if __has_builtin(__builtin_amdgcn_cvt_pk_bf16_f32)
  auto pk = __builtin_amdgcn_cvt_pk_bf16_f32(lo, hi);
  unsigned u;
  __builtin_memcpy(&u, &pk, sizeof(u));
  return u;
#else
  typedef __attribute__((ext_vector_type(2))) __bf16 v2bf;
  v2bf p;
  p[0] = (__bf16)lo;
  p[1] = (__bf16)hi;
  return __builtin_bit_cast(unsigned, p);
#endif
}

// A-operand fragment (16x32 bf16, M x K). ISA layout: lane holds row M=lane%16;
// elements 0..7 -> K = 8*half + e, elements 8..15 -> K = 16 + 8*half + (e-8).
__device__ __forceinline__ v16bf load_fragA(const unsigned short* p, int half) {
  FragU f;
  f.q[0] = *(const uint4*)(p + 8 * half);
  f.q[1] = *(const uint4*)(p + 16 + 8 * half);
  return f.v;
}

// B-operand fragment (32x16 bf16, K x N). ISA layout: lane holds column
// N=lane%16; elements 0..7 -> K = 8*half + e, elements 8..15 -> K = 16 + 8*half + (e-8).
__device__ __forceinline__ v16bf load_fragB(const unsigned short* p, int half) {
  FragU f;
  f.q[0] = *(const uint4*)(p + 8 * half);
  f.q[1] = *(const uint4*)(p + 16 + 8 * half);
  return f.v;
}

__device__ __forceinline__ v8f wmma_bf16(v16bf a, v16bf b, v8f c) {
  // (neg_a, A, neg_b, B, c_mod, C, reuse_a, reuse_b)
  return __builtin_amdgcn_wmma_f32_16x16x32_bf16(false, a, false, b,
                                                 (short)0, c, false, false);
}

// ---------------------------------------------------------------------------
// Kernel 1: fused Q/K/V projection. out = (x @ W^T + bias) * scale, bf16 out.
// Block tile 128(M) x 64(N), K-step 32, 8 waves of 32x32 each.
// ---------------------------------------------------------------------------
__global__ __launch_bounds__(256) void qkv_proj_kernel(
    const float* __restrict__ xq, const float* __restrict__ xk,
    const float* __restrict__ xv,
    const float* __restrict__ Wq, const float* __restrict__ bq,
    const float* __restrict__ Wk, const float* __restrict__ bk,
    const float* __restrict__ Wv, const float* __restrict__ bv,
    unsigned short* __restrict__ oq, unsigned short* __restrict__ okk,
    unsigned short* __restrict__ ov, float qscale)
{
  constexpr int BM = 128, BN = 64, BK = 32;
  __shared__ __align__(16) unsigned short Alds[BM * BK];  // 8 KB
  __shared__ __align__(16) unsigned short Blds[BN * BK];  // 4 KB

  const float* X; const float* W; const float* bias;
  unsigned short* O; float scale;
  if (blockIdx.z == 0)      { X = xq; W = Wq; bias = bq; O = oq;  scale = qscale; }
  else if (blockIdx.z == 1) { X = xk; W = Wk; bias = bk; O = okk; scale = 1.0f; }
  else                      { X = xv; W = Wv; bias = bv; O = ov;  scale = 1.0f; }

  const int t    = threadIdx.x;
  const int wave = t >> 5, lane = t & 31;
  const int half = lane >> 4, l16 = lane & 15;
  const int wm   = wave & 3, wn = wave >> 2;       // 4(M) x 2(N) wave grid
  const int m0   = blockIdx.y * BM, n0 = blockIdx.x * BN;

  v8f acc[2][2] = {};

  for (int k0 = 0; k0 < EMBED; k0 += BK) {
    __syncthreads();   // protect LDS from the previous iteration's readers
    // Stage A tile (128x32 fp32 -> bf16): 256 threads x 4 float4 loads.
    #pragma unroll
    for (int it = 0; it < 4; ++it) {
      int flat = it * 256 + t;
      int row = flat >> 3, cg = (flat & 7) << 2;
      float4 a = *(const float4*)(X + (size_t)(m0 + row) * EMBED + k0 + cg);
      uint2 pk;
      pk.x = pack2_bf16(a.x, a.y);
      pk.y = pack2_bf16(a.z, a.w);
      *(uint2*)&Alds[row * BK + cg] = pk;
    }
    // Stage B tile (64x32 of W, K-contiguous since W is [out,in]).
    #pragma unroll
    for (int it = 0; it < 2; ++it) {
      int flat = it * 256 + t;
      int row = flat >> 3, cg = (flat & 7) << 2;
      float4 w = *(const float4*)(W + (size_t)(n0 + row) * EMBED + k0 + cg);
      uint2 pk;
      pk.x = pack2_bf16(w.x, w.y);
      pk.y = pack2_bf16(w.z, w.w);
      *(uint2*)&Blds[row * BK + cg] = pk;
    }
    __syncthreads();

    v16bf a0 = load_fragA(&Alds[(wm * 32 +      l16) * BK], half);
    v16bf a1 = load_fragA(&Alds[(wm * 32 + 16 + l16) * BK], half);
    v16bf b0 = load_fragB(&Blds[(wn * 32 +      l16) * BK], half);
    v16bf b1 = load_fragB(&Blds[(wn * 32 + 16 + l16) * BK], half);
    acc[0][0] = wmma_bf16(a0, b0, acc[0][0]);
    acc[0][1] = wmma_bf16(a0, b1, acc[0][1]);
    acc[1][0] = wmma_bf16(a1, b0, acc[1][0]);
    acc[1][1] = wmma_bf16(a1, b1, acc[1][1]);
  }

  // Epilogue: bias + scale, store bf16. C layout: elem r -> M=r+8*half, N=l16.
  #pragma unroll
  for (int i = 0; i < 2; ++i) {
    #pragma unroll
    for (int j = 0; j < 2; ++j) {
      int ncol = n0 + wn * 32 + j * 16 + l16;
      float bb = bias[ncol];
      #pragma unroll
      for (int r = 0; r < 8; ++r) {
        int m = m0 + wm * 32 + i * 16 + r + 8 * half;
        float val = (acc[i][j][r] + bb) * scale;
        O[(size_t)m * EMBED + ncol] = bf16_one(val);
      }
    }
  }
}

// ---------------------------------------------------------------------------
// Kernel 2: masked softmax attention, flash style. One block = 128 q rows of
// one (b,h); 8 waves x 16 q rows. KV tiles of 64 staged in LDS.
// ---------------------------------------------------------------------------
__global__ __launch_bounds__(256) void attention_kernel(
    const unsigned short* __restrict__ qbf, const unsigned short* __restrict__ kbf,
    const unsigned short* __restrict__ vbf, const float* __restrict__ maskp,
    float* __restrict__ out)
{
  __shared__ __align__(16) unsigned short Klds[64 * 64];    // [key][d], 8 KB
  __shared__ __align__(16) unsigned short Vlds[64 * 64];    // [d][key], 8 KB
  __shared__ __align__(16) unsigned short Plds[8 * 16 * 64];// per-wave P, 16 KB

  const int t    = threadIdx.x;
  const int wave = t >> 5, lane = t & 31;
  const int half = lane >> 4, l16 = lane & 15;
  const int bh = blockIdx.y, b = bh >> 4, h = bh & 15;
  const int qw = blockIdx.x * 128 + wave * 16;

  // Q fragments for this wave's 16 rows (d = 0..31, 32..63). Scale pre-folded.
  const unsigned short* qrow =
      qbf + ((size_t)(b * SEQ + qw + l16)) * EMBED + h * HD;
  v16bf qf0 = load_fragA(qrow, half);
  v16bf qf1 = load_fragA(qrow + 32, half);

  v8f o[4] = {};
  float mrun[8], lrun[8];
  #pragma unroll
  for (int r = 0; r < 8; ++r) { mrun[r] = -3.0e38f; lrun[r] = 0.f; }

  for (int kv0 = 0; kv0 < SEQ; kv0 += 64) {
    __syncthreads();
    // Stage K tile [key][d] via async DMA to LDS; V transposed via VGPRs.
    #pragma unroll
    for (int it = 0; it < 4; ++it) {
      int flat = it * 256 + t;
      int key = flat >> 4, dg = (flat & 15) << 2;
      size_t goff = ((size_t)(b * SEQ + kv0 + key)) * EMBED + h * HD + dg;
#if USE_ASYNC_LDS
      // GVS mode: SGPR64 base + VGPR32 byte offset; VDST = LDS byte address
      // (low 32 bits of the generic shared pointer per the LDS aperture rule).
      unsigned ldsa = (unsigned)(size_t)&Klds[key * 64 + dg];
      int      voff = (int)(goff * sizeof(unsigned short));
      asm volatile("global_load_async_to_lds_b64 %0, %1, %2"
                   :: "v"(ldsa), "v"(voff), "s"(kbf)
                   : "memory");
#else
      *(uint2*)&Klds[key * 64 + dg] = *(const uint2*)(kbf + goff);
#endif
      uint2 vv = *(const uint2*)(vbf + goff);
      Vlds[(dg + 0) * 64 + key] = (unsigned short)(vv.x & 0xFFFFu);
      Vlds[(dg + 1) * 64 + key] = (unsigned short)(vv.x >> 16);
      Vlds[(dg + 2) * 64 + key] = (unsigned short)(vv.y & 0xFFFFu);
      Vlds[(dg + 3) * 64 + key] = (unsigned short)(vv.y >> 16);
    }
#if USE_ASYNC_LDS
    asm volatile("s_wait_asynccnt 0x0" ::: "memory");
#endif
    __syncthreads();

    if (kv0 + 64 < SEQ)  // pull the next K tile toward L2 (global_prefetch_b8)
      __builtin_prefetch(
          kbf + ((size_t)(b * SEQ + kv0 + 64 + (t & 63))) * EMBED + h * HD, 0, 1);

    // S = Q K^T : 4 column subtiles x (K=64 -> 2 chained WMMAs).
    v8f s[4];
    #pragma unroll
    for (int j = 0; j < 4; ++j) {
      const unsigned short* kb = &Klds[(j * 16 + l16) * 64];
      v8f a = {};
      a = wmma_bf16(qf0, load_fragB(kb, half), a);
      a = wmma_bf16(qf1, load_fragB(kb + 32, half), a);
      s[j] = a;
    }

    // Mask + online softmax in the C layout (row M = r + 8*half, col N = l16).
    const float* mbase = maskp + ((size_t)(b * SEQ) + qw) * SEQ + kv0 + l16;
    #pragma unroll
    for (int r = 0; r < 8; ++r) {
      const float* mr = mbase + (size_t)(r + 8 * half) * SEQ;
      float tmax = -3.0e38f;
      #pragma unroll
      for (int j = 0; j < 4; ++j) {
        float sm = s[j][r] - (1.0f - mr[j * 16]) * 1.0e6f;
        s[j][r] = sm;
        tmax = fmaxf(tmax, sm);
      }
      #pragma unroll
      for (int ml = 1; ml < 16; ml <<= 1)
        tmax = fmaxf(tmax, __shfl_xor(tmax, ml, 16));

      float mnew = fmaxf(mrun[r], tmax);
      float corr = __expf(mrun[r] - mnew);
      mrun[r] = mnew;
      float rs = 0.f;
      #pragma unroll
      for (int j = 0; j < 4; ++j) {
        float p = __expf(s[j][r] - mnew);
        s[j][r] = p;
        rs += p;
      }
      #pragma unroll
      for (int ml = 1; ml < 16; ml <<= 1)
        rs += __shfl_xor(rs, ml, 16);
      lrun[r] = lrun[r] * corr + rs;
      #pragma unroll
      for (int n = 0; n < 4; ++n) o[n][r] *= corr;
    }

    // Relayout P (C layout) -> A-fragment layout via per-wave LDS scratch.
    unsigned short* pw = Plds + wave * (16 * 64);
    #pragma unroll
    for (int r = 0; r < 8; ++r)
      #pragma unroll
      for (int j = 0; j < 4; ++j)
        pw[(r + 8 * half) * 64 + j * 16 + l16] = bf16_one(s[j][r]);
    asm volatile("s_wait_dscnt 0x0" ::: "memory");  // wave-internal LDS RAW

    const unsigned short* pr = pw + l16 * 64;
    v16bf p0 = load_fragA(pr, half);        // keys 0..31 of this tile
    v16bf p1 = load_fragA(pr + 32, half);   // keys 32..63
    #pragma unroll
    for (int n = 0; n < 4; ++n) {
      const unsigned short* vb = &Vlds[(n * 16 + l16) * 64];
      o[n] = wmma_bf16(p0, load_fragB(vb, half), o[n]);
      o[n] = wmma_bf16(p1, load_fragB(vb + 32, half), o[n]);
    }
  }

  // Normalize (one reciprocal per row) and write fp32 output [B,S,H*d].
  #pragma unroll
  for (int r = 0; r < 8; ++r) {
    float inv = 1.0f / lrun[r];
    size_t rowoff =
        ((size_t)(b * SEQ) + qw + r + 8 * half) * EMBED + h * HD + l16;
    #pragma unroll
    for (int n = 0; n < 4; ++n)
      out[rowoff + n * 16] = o[n][r] * inv;
  }
}

// ---------------------------------------------------------------------------
extern "C" void kernel_launch(void* const* d_in, const int* in_sizes, int n_in,
                              void* d_out, int out_size, void* d_ws, size_t ws_size,
                              hipStream_t stream) {
  (void)in_sizes; (void)n_in; (void)out_size; (void)ws_size;
  const float* xq = (const float*)d_in[0];
  const float* xk = (const float*)d_in[1];
  const float* xv = (const float*)d_in[2];
  const float* mk = (const float*)d_in[3];
  const float* Wq = (const float*)d_in[4];
  const float* bq = (const float*)d_in[5];
  const float* Wk = (const float*)d_in[6];
  const float* bk = (const float*)d_in[7];
  const float* Wv = (const float*)d_in[8];
  const float* bv = (const float*)d_in[9];

  // Workspace: Q, K, V as bf16 [B*S, E] = 8 MB each (24 MB total).
  unsigned short* qbf = (unsigned short*)d_ws;
  unsigned short* kbf = qbf + (size_t)BATCH * SEQ * EMBED;
  unsigned short* vbf = kbf + (size_t)BATCH * SEQ * EMBED;

  dim3 gp(EMBED / 64, (BATCH * SEQ) / 128, 3);
  qkv_proj_kernel<<<gp, dim3(256), 0, stream>>>(
      xq, xk, xv, Wq, bq, Wk, bk, Wv, bv, qbf, kbf, vbf, 0.125f /* 1/sqrt(64) */);

  dim3 ga(SEQ / 128, BATCH * HEADS);
  attention_kernel<<<ga, dim3(256), 0, stream>>>(qbf, kbf, vbf, mk, (float*)d_out);
}
